// MYGKT2015_88338887344576
// MI455X (gfx1250) — matrix operand order repaired and checked by reading, not compiled, not run on hardware
//
#include <hip/hip_runtime.h>
#include <hip/hip_bf16.h>

#define BATCH 16
#define SEQ   64
#define NQ    256
#define HD    64
#define H2    128
#define H3    192
#define HF_T  (NQ * HD)            // 16384: stride of one time slice of h_full
#define HF_B  ((SEQ + 1) * HF_T)   // 1064960: stride of one batch slice of h_full

typedef __attribute__((ext_vector_type(16))) __bf16 v16bf;
typedef __attribute__((ext_vector_type(8)))  __bf16 v8bf;
typedef __attribute__((ext_vector_type(8)))  float  v8f;

// packed-weight fragment table offsets (in fragments of 512 bf16 = 1KB)
#define FRAG_W1   0     // 16 frags: kb(4) x nt(4)
#define FRAG_W2   16    //  8 frags: kb(2) x nt(4)
#define FRAG_WIH  24    // 48 frags: kb(4) x nt(12)
#define FRAG_WHH  72    // 24 frags: kb(2) x nt(12)
#define FRAG_QE   96    // 32 frags: kb(2) x tn(16)
#define NFRAGS    128

__device__ __forceinline__ v8f wmma_bf16(v16bf a, v16bf b, v8f c) {
  return __builtin_amdgcn_wmma_f32_16x16x32_bf16(false, a, false, b, (short)0, c,
                                                 false, false);
}

// Load one pre-packed B fragment: lane-contiguous 32B -> two b128 loads, no cvt
__device__ __forceinline__ v16bf load_frag(const __bf16* __restrict__ wp, int f) {
  const int lane = threadIdx.x & 31;
  return *((const v16bf*)(wp + f * 512) + lane);
}

// Build A fragment (16x32) from a per-lane row pointer using 4 aligned float4 loads.
// 16-bit A layout: elems 0..7 <- row[k0 + 8*half + 0..7]; 8..15 <- row[k0+16+8*half+0..7]
__device__ __forceinline__ v16bf afrag_row(const float* __restrict__ row, int k0) {
  const int half = (threadIdx.x >> 4) & 1;
  const float4 c0 = *(const float4*)(row + k0 + 8 * half);
  const float4 c1 = *(const float4*)(row + k0 + 8 * half + 4);
  const float4 c2 = *(const float4*)(row + k0 + 16 + 8 * half);
  const float4 c3 = *(const float4*)(row + k0 + 16 + 8 * half + 4);
  v16bf a;
  a[0]=(__bf16)c0.x;  a[1]=(__bf16)c0.y;  a[2]=(__bf16)c0.z;  a[3]=(__bf16)c0.w;
  a[4]=(__bf16)c1.x;  a[5]=(__bf16)c1.y;  a[6]=(__bf16)c1.z;  a[7]=(__bf16)c1.w;
  a[8]=(__bf16)c2.x;  a[9]=(__bf16)c2.y;  a[10]=(__bf16)c2.z; a[11]=(__bf16)c2.w;
  a[12]=(__bf16)c3.x; a[13]=(__bf16)c3.y; a[14]=(__bf16)c3.z; a[15]=(__bf16)c3.w;
  return a;
}

// ---- one-time: pack all weight matrices into bf16 B-fragment layout ----
// Fragment f: lane l holds B[k0+8h+(j&7)+16*(j>>3)][n0+l%16], h=l/16, j=0..15 (32B/lane).
__global__ void k_pack(const float* __restrict__ w1, const float* __restrict__ w2,
                       const float* __restrict__ w_ih, const float* __restrict__ w_hh,
                       const float* __restrict__ q_emb, __bf16* __restrict__ dst) {
  const int f = blockIdx.x;            // 0..127
  const int lane = threadIdx.x & 31;   // column n0 + lane%16; lane/16 picks k half
  v16bf o;
  if (f < FRAG_W2) {                       // w1 (128x64), B[k][n] = w1[k*64+n]
    const int kb = f >> 2, nt = f & 3;
#pragma unroll
    for (int j = 0; j < 16; ++j) o[j] = (__bf16)w1[(kb * 32 + (j & 7) + 8 * (lane >> 4) + 16 * (j >> 3)) * HD + nt * 16 + (lane & 15)];
  } else if (f < FRAG_WIH) {               // w2 (64x64), B[k][n] = w2[k*64+n]
    const int g = f - FRAG_W2, kb = g >> 2, nt = g & 3;
#pragma unroll
    for (int j = 0; j < 16; ++j) o[j] = (__bf16)w2[(kb * 32 + (j & 7) + 8 * (lane >> 4) + 16 * (j >> 3)) * HD + nt * 16 + (lane & 15)];
  } else if (f < FRAG_WHH) {               // w_ih^T: B[k][n] = w_ih[n*128+k]
    const int g = f - FRAG_WIH, kb = g / 12, nt = g % 12;
#pragma unroll
    for (int j = 0; j < 16; ++j) o[j] = (__bf16)w_ih[(nt * 16 + (lane & 15)) * H2 + kb * 32 + (j & 7) + 8 * (lane >> 4) + 16 * (j >> 3)];
  } else if (f < FRAG_QE) {                // w_hh^T: B[k][n] = w_hh[n*64+k]
    const int g = f - FRAG_WHH, kb = g / 12, nt = g % 12;
#pragma unroll
    for (int j = 0; j < 16; ++j) o[j] = (__bf16)w_hh[(nt * 16 + (lane & 15)) * HD + kb * 32 + (j & 7) + 8 * (lane >> 4) + 16 * (j >> 3)];
  } else {                                 // q_emb^T: B[k][c] = q_emb[c*64+k]
    const int g = f - FRAG_QE, kb = g >> 4, tn = g & 15;
#pragma unroll
    for (int j = 0; j < 16; ++j) o[j] = (__bf16)q_emb[(tn * 16 + (lane & 15)) * HD + kb * 32 + (j & 7) + 8 * (lane >> 4) + 16 * (j >> 3)];
  }
  *((v16bf*)(dst + f * 512) + lane) = o;
}

// ---------------- broadcast init_h into h_full[:, 0] ----------------
__global__ void k_init_h0(const float* __restrict__ init_h, float* __restrict__ hfull) {
  const int i = blockIdx.x * 256 + threadIdx.x;   // exactly B*HF_T threads
  const int b = i >> 14;
  const int rest = i & (HF_T - 1);
  hfull[b * HF_B + rest] = init_h[rest];
}

// ---------------- sim[bt, c] = q_emb[q[bt]] . q_emb[c]  (WMMA) --------------
__global__ void k_sim(const int* __restrict__ q, const float* __restrict__ q_emb,
                      const __bf16* __restrict__ wpack, float* __restrict__ sim) {
  const int tm = blockIdx.x;            // 64 row tiles (B*SEQ rows)
  const int tn = blockIdx.y;            // 16 col tiles (NQ cols)
  const int lane = threadIdx.x & 31;
  const int m = lane & 15, half = lane >> 4;
  const int qi = q[tm * 16 + m];
  const float* arow = q_emb + qi * HD;
  v8f acc = {};
#pragma unroll
  for (int kb = 0; kb < 2; ++kb)
    acc = wmma_bf16(afrag_row(arow, kb * 32),
                    load_frag(wpack, FRAG_QE + kb * 16 + tn), acc);
#pragma unroll
  for (int v = 0; v < 8; ++v)
    sim[(tm * 16 + v + 8 * half) * NQ + tn * 16 + (lane & 15)] = acc[v];
}

// -------- per t: fused tail-select + 2-layer MLP + gi GEMM ------------------
// One wave per 16-row tile (rows = b*NQ + c). gi written in [c][gate][col][b]
// layout so the sequential GRU scan reads contiguously.
__global__ void k_mlp_gi(int t, const int* __restrict__ q, const int* __restrict__ r,
                         const float* __restrict__ x_emb, const float* __restrict__ q_emb,
                         const __bf16* __restrict__ wpack,
                         const float* __restrict__ b1, const float* __restrict__ b2,
                         const float* __restrict__ b_ih,
                         const float* __restrict__ hfull, float* __restrict__ gi2) {
  __shared__ float lds[2][16][HD + 4];
  const int tile = blockIdx.x;          // 0..255
  const int lane = threadIdx.x & 31;
  const int m = lane & 15, half = lane >> 4;
  const int row = tile * 16 + m;        // = b*NQ + c
  const int b = tile >> 4;              // 16 | NQ: whole tile shares b
  const int c = row & 255;
  const float* hrow = hfull + b * HF_B + t * HF_T + c * HD;
  const int qbt = q[b * SEQ + t];
  const float* tail = (c == qbt) ? (x_emb + (qbt + NQ * r[b * SEQ + t]) * HD)
                                 : (q_emb + c * HD);
  v16bf a1[4];
  a1[0] = afrag_row(hrow, 0);  a1[1] = afrag_row(hrow, 32);
  a1[2] = afrag_row(tail, 0);  a1[3] = afrag_row(tail, 32);
  // layer 1: u = relu(A1 @ w1 + b1) -> lds[0]
#pragma unroll
  for (int nt = 0; nt < 4; ++nt) {
    v8f acc = {};
#pragma unroll
    for (int kb = 0; kb < 4; ++kb)
      acc = wmma_bf16(a1[kb], load_frag(wpack, FRAG_W1 + kb * 4 + nt), acc);
    const int n = nt * 16 + (lane & 15);
    const float bb = b1[n];
#pragma unroll
    for (int v = 0; v < 8; ++v) lds[0][v + 8 * half][n] = fmaxf(acc[v] + bb, 0.f);
  }
  __syncthreads();
  // layer 2: mvec = U @ w2 + b2 -> lds[1]
  v16bf ua[2];
  ua[0] = afrag_row(&lds[0][m][0], 0);
  ua[1] = afrag_row(&lds[0][m][0], 32);
#pragma unroll
  for (int nt = 0; nt < 4; ++nt) {
    v8f acc = {};
#pragma unroll
    for (int kb = 0; kb < 2; ++kb)
      acc = wmma_bf16(ua[kb], load_frag(wpack, FRAG_W2 + kb * 4 + nt), acc);
    const int n = nt * 16 + (lane & 15);
    const float bb = b2[n];
#pragma unroll
    for (int v = 0; v < 8; ++v) lds[1][v + 8 * half][n] = acc[v] + bb;
  }
  __syncthreads();
  // gi = [mvec | h] @ w_ih^T + b_ih ; reuse a1[0..1] as the h half of A2
  v16bf a2[4];
  a2[0] = afrag_row(&lds[1][m][0], 0);
  a2[1] = afrag_row(&lds[1][m][0], 32);
  a2[2] = a1[0];
  a2[3] = a1[1];
#pragma unroll
  for (int nt = 0; nt < 12; ++nt) {
    v8f acc = {};
#pragma unroll
    for (int kb = 0; kb < 4; ++kb)
      acc = wmma_bf16(a2[kb], load_frag(wpack, FRAG_WIH + kb * 12 + nt), acc);
    const int ng = nt * 16 + (lane & 15);
    const int g = ng >> 6, col = ng & 63;
    const float bb = b_ih[ng];
#pragma unroll
    for (int v = 0; v < 8; ++v) {
      const int cc = (tile & 15) * 16 + v + 8 * half;      // concept index
      gi2[(((cc * 3) + g) * 64 + col) * 16 + b] = acc[v] + bb;
    }
  }
}

// -------- per t: sequential GRU scan over concept axis ----------------------
// one workgroup, 4 waves; wave w owns columns [16w,16w+16) of all 3 gates.
// State: hT f32 [col][b] (contig 32B read/write), hbf bf16 mirror for A-frags.
__global__ void k_gru_scan(int t, const __bf16* __restrict__ wpack,
                           const float* __restrict__ b_hh,
                           const float* __restrict__ gi2, float* __restrict__ hfull) {
  __shared__ float  hT[2][HD][BATCH];
  __shared__ __bf16 hbf[2][BATCH][72];
  const int tid = threadIdx.x;          // 0..127
  const int w = tid >> 5;
  const int lane = tid & 31;
  const int m = lane & 15, half = lane >> 4;
  const int col = w * 16 + m;
  v16bf bhh[3][2];
#pragma unroll
  for (int g = 0; g < 3; ++g)
#pragma unroll
    for (int kb = 0; kb < 2; ++kb)
      bhh[g][kb] = load_frag(wpack, FRAG_WHH + kb * 12 + g * 4 + w);
  const float bhr = b_hh[col], bhz = b_hh[HD + col], bhn = b_hh[2 * HD + col];
  for (int i = tid; i < HD * BATCH; i += 128) hT[0][i >> 4][i & 15] = 0.f;
  for (int i = tid; i < BATCH * 72; i += 128) hbf[0][i / 72][i % 72] = (__bf16)0.f;
  __syncthreads();
  int p = 0;
  float* hout = hfull + (t + 1) * HF_T;
  for (int c = 0; c < NQ; ++c) {
    if (c + 8 < NQ)  // hide gi latency behind the serial chain
      __builtin_prefetch(gi2 + (c + 8) * 3072 + tid * 24, 0, 1);
    // A fragments from bf16 mirror: 4 aligned 16B LDS loads, no cvt
    v16bf ha[2];
#pragma unroll
    for (int kb = 0; kb < 2; ++kb) {
      const __bf16* rp = &hbf[p][m][kb * 32 + 8 * half];
      const v8bf lo = *(const v8bf*)rp;
      const v8bf hi = *(const v8bf*)(rp + 16);
      ha[kb] = __builtin_shufflevector(lo, hi, 0, 1, 2, 3, 4, 5, 6, 7,
                                       8, 9, 10, 11, 12, 13, 14, 15);
    }
    v8f gr = {}, gz = {}, gn = {};
#pragma unroll
    for (int kb = 0; kb < 2; ++kb) {
      gr = wmma_bf16(ha[kb], bhh[0][kb], gr);
      gz = wmma_bf16(ha[kb], bhh[1][kb], gz);
      gn = wmma_bf16(ha[kb], bhh[2][kb], gn);
    }
    // gate inputs: contiguous float4 pairs thanks to [c][g][col][b] layout
    const float* gci = gi2 + c * 3072;
    float ir[8], iz[8], inn[8], hp[8], hn[8];
    ((float4*)ir)[0]  = *(const float4*)(gci + (0 * 64 + col) * 16 + 8 * half);
    ((float4*)ir)[1]  = *(const float4*)(gci + (0 * 64 + col) * 16 + 8 * half + 4);
    ((float4*)iz)[0]  = *(const float4*)(gci + (1 * 64 + col) * 16 + 8 * half);
    ((float4*)iz)[1]  = *(const float4*)(gci + (1 * 64 + col) * 16 + 8 * half + 4);
    ((float4*)inn)[0] = *(const float4*)(gci + (2 * 64 + col) * 16 + 8 * half);
    ((float4*)inn)[1] = *(const float4*)(gci + (2 * 64 + col) * 16 + 8 * half + 4);
    ((float4*)hp)[0]  = *(const float4*)(&hT[p][col][8 * half]);
    ((float4*)hp)[1]  = *(const float4*)(&hT[p][col][8 * half + 4]);
#pragma unroll
    for (int v = 0; v < 8; ++v) {
      const float rg = 1.f / (1.f + __expf(-(ir[v] + gr[v] + bhr)));
      const float zg = 1.f / (1.f + __expf(-(iz[v] + gz[v] + bhz)));
      const float ng = tanhf(inn[v] + rg * (gn[v] + bhn));
      hn[v] = (1.f - zg) * ng + zg * hp[v];
    }
    *(float4*)(&hT[1 - p][col][8 * half])     = ((float4*)hn)[0];
    *(float4*)(&hT[1 - p][col][8 * half + 4]) = ((float4*)hn)[1];
#pragma unroll
    for (int v = 0; v < 8; ++v) {
      const int b = 8 * half + v;
      hbf[1 - p][b][col] = (__bf16)hn[v];
      hout[b * HF_B + c * HD + col] = hn[v];
    }
    __syncthreads();
    p ^= 1;
  }
}

// -------- hw[b,u,c] = h_full[b,u,c,:] . out_w -------------------------------
__global__ void k_hw(const float* __restrict__ hfull, const float* __restrict__ out_w,
                     float* __restrict__ hw) {
  const int i = blockIdx.x * 256 + threadIdx.x;   // exactly B*(SEQ+1)*NQ threads
  const int b = i / ((SEQ + 1) * NQ);
  const int rest = i % ((SEQ + 1) * NQ);          // = u*NQ + c
  const float* h = hfull + b * HF_B + rest * HD;
  float s = 0.f;
#pragma unroll
  for (int k = 0; k < HD; ++k) s += h[k] * out_w[k];
  hw[i] = s;
}

// -------- causal decayed softmax + logits + sigmoid -------------------------
__global__ void k_attn(const float* __restrict__ sim, const float* __restrict__ hw,
                       const float* __restrict__ bias, const float* __restrict__ theta,
                       float* __restrict__ y) {
  const int bt = blockIdx.x;            // b*SEQ + t
  const int b = bt >> 6, t = bt & 63;
  const int c = threadIdx.x;            // 0..255
  const float rate = __expf(theta[0]);
  const float* simb = sim + b * SEQ * NQ + c;
  const float* hwb = hw + b * (SEQ + 1) * NQ + c;
  float mx = -3.402823466e38f;
  for (int u = 0; u <= t; ++u) {
    const float s = __expf(-rate * (float)(t - u)) * simb[u * NQ];
    mx = fmaxf(mx, s);
  }
  float den = 0.f, num = 0.f;
  for (int u = 0; u <= t; ++u) {
    const float s = __expf(-rate * (float)(t - u)) * simb[u * NQ];
    const float e = __expf(s - mx);
    den += e;
    num += e * hwb[u * NQ];
  }
  const float logit = hwb[(t + 1) * NQ] + num / den + bias[c];
  y[bt * NQ + c] = 1.f / (1.f + __expf(-logit));
}

extern "C" void kernel_launch(void* const* d_in, const int* in_sizes, int n_in,
                              void* d_out, int out_size, void* d_ws, size_t ws_size,
                              hipStream_t stream) {
  (void)in_sizes; (void)n_in; (void)out_size; (void)ws_size;
  const int*   q      = (const int*)d_in[0];
  const int*   r      = (const int*)d_in[1];
  const float* x_emb  = (const float*)d_in[2];
  const float* q_emb  = (const float*)d_in[3];
  const float* init_h = (const float*)d_in[4];
  const float* w1     = (const float*)d_in[5];
  const float* b1     = (const float*)d_in[6];
  const float* w2     = (const float*)d_in[7];
  const float* b2     = (const float*)d_in[8];
  const float* w_ih   = (const float*)d_in[9];
  const float* w_hh   = (const float*)d_in[10];
  const float* b_ih   = (const float*)d_in[11];
  const float* b_hh   = (const float*)d_in[12];
  const float* bias   = (const float*)d_in[13];
  const float* out_w  = (const float*)d_in[14];
  const float* theta  = (const float*)d_in[15];

  float* y     = (float*)d_out;                 // (B, N, NQ)
  float* hfull = y + BATCH * SEQ * NQ;          // (B, N+1, NQ, H)

  float*  sim   = (float*)d_ws;                         // 262144 f32
  float*  hw    = sim + BATCH * SEQ * NQ;               // 266240 f32
  float*  gi2   = hw + BATCH * (SEQ + 1) * NQ;          // 786432 f32 [c][g][col][b]
  __bf16* wpack = (__bf16*)(gi2 + BATCH * NQ * H3);     // 128 frags * 1KB

  k_pack<<<dim3(NFRAGS), dim3(32), 0, stream>>>(w1, w2, w_ih, w_hh, q_emb, wpack);
  k_init_h0<<<dim3(1024), dim3(256), 0, stream>>>(init_h, hfull);
  k_sim<<<dim3(64, 16), dim3(32), 0, stream>>>(q, q_emb, wpack, sim);
  for (int t = 0; t < SEQ; ++t) {
    k_mlp_gi<<<dim3(256), dim3(32), 0, stream>>>(t, q, r, x_emb, q_emb, wpack,
                                                 b1, b2, b_ih, hfull, gi2);
    k_gru_scan<<<dim3(1), dim3(128), 0, stream>>>(t, wpack, b_hh, gi2, hfull);
  }
  k_hw<<<dim3(1040), dim3(256), 0, stream>>>(hfull, out_w, hw);
  k_attn<<<dim3(1024), dim3(256), 0, stream>>>(sim, hw, bias, theta, y);
}
